// SimpleQuantizer_45956150067818
// MI455X (gfx1250) — matrix operand order repaired and checked by reading, not compiled, not run on hardware
//
#include <hip/hip_runtime.h>

// ---------------------------------------------------------------------------
// VQ-VAE quantizer for MI455X (gfx1250, wave32, WMMA + Tensor Data Mover).
//   scores = z @ W^T  (bf16 WMMA, fp32 accum), argmax_k(score_k - 0.5*||w_k||^2)
//   out    = [ W[idx] (fp32) | idx (as float) | 1.25 * mean ||W[idx]-z||^2 ]
// Codebook (bf16) streamed through double-buffered LDS by the TDM while all
// eight waves run two independent WMMA accumulation chains.
// ---------------------------------------------------------------------------

typedef __attribute__((ext_vector_type(16))) __bf16         v16bf;
typedef __attribute__((ext_vector_type(8)))  float          v8f;
typedef __attribute__((ext_vector_type(8)))  unsigned short v8us;
typedef __attribute__((ext_vector_type(16))) unsigned short v16us;
typedef __attribute__((ext_vector_type(4)))  unsigned int   v4u;
typedef __attribute__((ext_vector_type(4)))  int            v4i;
typedef __attribute__((ext_vector_type(8)))  int            v8i;

#define DIMD          256
#define KCODES        1024
#define NROWS         65536          // 32 * 2048
#define ROWS_PER_WAVE 16
#define WAVES_PER_WG  8
#define ROWS_PER_WG   (ROWS_PER_WAVE * WAVES_PER_WG)   // 128
#define CHUNK         32             // codes per LDS buffer
#define NTILES        (CHUNK / 16)   // 2
#define NCHUNKS       (KCODES / CHUNK)                 // 32
#define PITCH_US      264            // ushorts per LDS code row (528 B, bank-skewed)

// ---- Tensor Data Mover availability (guarded: fall back to coop loads) ----
#if defined(__HIP_DEVICE_COMPILE__)
#  if __has_builtin(__builtin_amdgcn_tensor_load_to_lds) && \
      __has_builtin(__builtin_amdgcn_s_wait_tensorcnt)
#    define VQ_USE_TDM 1
#  endif
#endif
#ifndef VQ_USE_TDM
#  define VQ_USE_TDM 0
#endif

static __device__ __forceinline__ unsigned short f2bf(float f) {
  // round-to-nearest-even fp32 -> bf16 bits
  unsigned u = __builtin_bit_cast(unsigned, f);
  u = u + 0x7FFFu + ((u >> 16) & 1u);
  return (unsigned short)(u >> 16);
}

#if VQ_USE_TDM
// Issue one TDM descriptor: copy CHUNK*256 bf16 (16 KB, contiguous in global)
// into LDS at lds_off, inserting 4 DWORDs of pad every 128 DWORDs so rows land
// on the 528-byte bank-skewed pitch. Wave-uniform; call from a single wave.
static __device__ __forceinline__ void tdm_load_chunk(
    const unsigned short* gsrc, unsigned lds_off) {
  const unsigned long long ga = (unsigned long long)(size_t)gsrc;
  v4u g0;
  g0[0] = 1u;                                            // count=1, user mode
  g0[1] = lds_off;                                       // lds_addr (bytes)
  g0[2] = (unsigned)(ga & 0xFFFFFFFFu);                  // global_addr[31:0]
  g0[3] = (unsigned)((ga >> 32) & 0x01FFFFFFu)           // global_addr[56:32]
          | 0x80000000u;                                 // type = 2 ("image")
  v8i g1;
  g1[0] = (int)((1u << 16)      // data_size = 1 (2-byte elements)
                | (1u << 20)    // pad_enable
                | (6u << 22)    // pad_interval: every 128 DWORDs (512 B row)
                | (3u << 25));  // pad_amount: 4 DWORDs (16 B skew)
  g1[1] = 0;                    // atomic barrier addr; tensor_dim0[15:0] = 0
  g1[2] = (int)(0x4u            // tensor_dim0[31:16] -> dim0 = 262144 elems
                | (1u << 16));  // tensor_dim1[15:0] = 1
  g1[3] = (int)(0x2000u << 16); // tile_dim0 = 8192 elems (CHUNK*256)
  g1[4] = 1;                    // tile_dim1 = 1, tile_dim2 = 0
  g1[5] = (int)0x40000u;        // tensor_dim0_stride = 262144
  g1[6] = 0;
  g1[7] = 0;
  const v4i z4 = {0, 0, 0, 0};
#if __clang_major__ >= 23
  const v8i z8 = {0, 0, 0, 0, 0, 0, 0, 0};
  __builtin_amdgcn_tensor_load_to_lds(g0, g1, z4, z4, z8, 0);
#else
  __builtin_amdgcn_tensor_load_to_lds(g0, g1, z4, z4, 0);
#endif
}
#endif  // VQ_USE_TDM

// --------------------------- prep kernel -----------------------------------
__global__ void vq_prep(const float* __restrict__ W,
                        unsigned short* __restrict__ Wb,
                        float* __restrict__ bias,
                        float* __restrict__ lossSlot) {
  const int k = blockIdx.x;
  const int t = threadIdx.x;            // 256 threads = DIMD
  const float wval = W[(size_t)k * DIMD + t];
  Wb[(size_t)k * DIMD + t] = f2bf(wval);

  float s = wval * wval;
#pragma unroll
  for (int off = 1; off < 32; off <<= 1) s += __shfl_xor(s, off, 32);

  __shared__ float ps[WAVES_PER_WG];
  if ((t & 31) == 0) ps[t >> 5] = s;
  __syncthreads();
  if (t == 0) {
    float tot = 0.f;
#pragma unroll
    for (int i = 0; i < WAVES_PER_WG; i++) tot += ps[i];
    bias[k] = -0.5f * tot;
    if (k == 0) *lossSlot = 0.f;
  }
}

// --------------------------- main kernel -----------------------------------
__global__ void __launch_bounds__(256) vq_main(
    const float* __restrict__ z, const float* __restrict__ W,
    const unsigned short* __restrict__ Wb, const float* __restrict__ bias,
    float* __restrict__ outZ, float* __restrict__ outIdx,
    float* __restrict__ outLoss) {
  __shared__ unsigned short lB[2][CHUNK * PITCH_US];   // 2 x 16.5 KB
  __shared__ int sIdx[WAVES_PER_WG * ROWS_PER_WAVE];

  const int tid  = threadIdx.x;
  const int lane = tid & 31;
  const int wv   = tid >> 5;       // wave id in workgroup
  const int m    = lane & 15;
  const int hv   = lane >> 4;      // 16-lane half

  const long rowbase = (long)blockIdx.x * ROWS_PER_WG + (long)wv * ROWS_PER_WAVE;
  const long arow    = rowbase + m;
  const float* zr    = z + arow * DIMD;

  // ---- Load this wave's 16 z-rows once, convert to bf16 A fragments.
  // ISA 16-bit A 16x32 layout: half 0 -> K = {0..7, 16..23}, half 1 -> {8..15, 24..31}.
  v16bf a[8];
#pragma unroll
  for (int ks = 0; ks < 8; ks++) {
    const int base  = 32 * ks + 8 * hv;
    const float4 c0 = *(const float4*)(zr + base);
    const float4 c1 = *(const float4*)(zr + base + 4);
    const float4 c2 = *(const float4*)(zr + base + 16);
    const float4 c3 = *(const float4*)(zr + base + 20);
    v16us ua;
    ua[0]  = f2bf(c0.x); ua[1]  = f2bf(c0.y); ua[2]  = f2bf(c0.z); ua[3]  = f2bf(c0.w);
    ua[4]  = f2bf(c1.x); ua[5]  = f2bf(c1.y); ua[6]  = f2bf(c1.z); ua[7]  = f2bf(c1.w);
    ua[8]  = f2bf(c2.x); ua[9]  = f2bf(c2.y); ua[10] = f2bf(c2.z); ua[11] = f2bf(c2.w);
    ua[12] = f2bf(c3.x); ua[13] = f2bf(c3.y); ua[14] = f2bf(c3.z); ua[15] = f2bf(c3.w);
    a[ks] = __builtin_bit_cast(v16bf, ua);
  }

  float best[8];
  int   besti[8];
#pragma unroll
  for (int v = 0; v < 8; v++) { best[v] = -3.4e38f; besti[v] = 0; }

#if VQ_USE_TDM
  const unsigned ldsOff[2] = {
      (unsigned)(unsigned long long)(size_t)(&lB[0][0]),
      (unsigned)(unsigned long long)(size_t)(&lB[1][0])};
  // Prologue: DMA chunk 0 into buffer 0.
  if (wv == 0) {
    tdm_load_chunk(Wb, ldsOff[0]);
    __builtin_amdgcn_s_wait_tensorcnt(0);
  }
  __syncthreads();
#else
  // Prologue: cooperative stage of chunk 0 into buffer 0.
#pragma unroll
  for (int u0 = 0; u0 < (CHUNK * 32) / 256; u0++) {
    const int u = u0 * 256 + tid;
    const int r = u >> 5;
    const int o = u & 31;
    const uint4 d = *(const uint4*)(Wb + (size_t)r * DIMD + o * 8);
    *(uint4*)(&lB[0][r * PITCH_US + o * 8]) = d;
  }
  __syncthreads();
#endif

  int cur = 0;
  for (int ch = 0; ch < NCHUNKS; ch++) {
    const int cb = ch * CHUNK;

    // ---- kick off the next chunk into the other buffer
#if VQ_USE_TDM
    if (ch + 1 < NCHUNKS && wv == 0)
      tdm_load_chunk(Wb + (size_t)(cb + CHUNK) * DIMD, ldsOff[cur ^ 1]);
#else
    if (ch + 1 < NCHUNKS) {
#pragma unroll
      for (int u0 = 0; u0 < (CHUNK * 32) / 256; u0++) {
        const int u = u0 * 256 + tid;
        const int r = u >> 5;
        const int o = u & 31;
        const uint4 d = *(const uint4*)(Wb + (size_t)(cb + CHUNK + r) * DIMD + o * 8);
        *(uint4*)(&lB[cur ^ 1][r * PITCH_US + o * 8]) = d;
      }
    }
#endif

    // ---- compute on current buffer
#pragma unroll
    for (int nt = 0; nt < NTILES; nt++) {
      const int c = cb + nt * 16 + m;                 // this lane's code column
      const unsigned short* bp = &lB[cur][(nt * 16 + m) * PITCH_US];
      // Preload all eight B fragments into distinct registers so one
      // s_wait_dscnt covers the whole tile.
      v16us b[8];
#pragma unroll
      for (int ks = 0; ks < 8; ks++) {
        // ISA 16-bit B 32x16 layout: half 0 -> K = {0..7, 16..23}, half 1 -> {8..15, 24..31}
        const v8us b0 = *(const v8us*)(bp + 32 * ks + 8 * hv);
        const v8us b1 = *(const v8us*)(bp + 32 * ks + 8 * hv + 16);
        b[ks] = __builtin_shufflevector(
            b0, b1, 0, 1, 2, 3, 4, 5, 6, 7, 8, 9, 10, 11, 12, 13, 14, 15);
      }
      // Two independent accumulation chains (break the WMMA RAW chain).
      v8f acc0 = {};
      v8f acc1 = {};
#pragma unroll
      for (int ks = 0; ks < 4; ks++) {
        acc0 = __builtin_amdgcn_wmma_f32_16x16x32_bf16(
            false, a[2 * ks], false, __builtin_bit_cast(v16bf, b[2 * ks]),
            (short)0, acc0, false, false);
        acc1 = __builtin_amdgcn_wmma_f32_16x16x32_bf16(
            false, a[2 * ks + 1], false, __builtin_bit_cast(v16bf, b[2 * ks + 1]),
            (short)0, acc1, false, false);
      }
      const float bc = bias[c];
#pragma unroll
      for (int v = 0; v < 8; v++) {
        const float s = acc0[v] + acc1[v] + bc;
        if (s > best[v]) { best[v] = s; besti[v] = c; }
      }
    }

#if VQ_USE_TDM
    if (wv == 0) __builtin_amdgcn_s_wait_tensorcnt(0);  // next chunk landed
#endif
    __syncthreads();
    cur ^= 1;
  }

  // ---- argmax reduction across each 16-lane half (C/D layout: lanes 0-15
  //      hold rows 0..7, lanes 16-31 hold rows 8..15). Tie-break low index.
#pragma unroll
  for (int v = 0; v < 8; v++) {
#pragma unroll
    for (int off = 1; off < 16; off <<= 1) {
      const float ob = __shfl_xor(best[v], off, 32);
      const int   oi = __shfl_xor(besti[v], off, 32);
      if (ob > best[v] || (ob == best[v] && oi < besti[v])) {
        best[v] = ob; besti[v] = oi;
      }
    }
  }
  if (m == 0) {
#pragma unroll
    for (int v = 0; v < 8; v++) sIdx[wv * ROWS_PER_WAVE + hv * 8 + v] = besti[v];
  }
  __syncthreads();

  // ---- Gather fp32 codebook rows to output + fp32 commitment loss.
  float lacc = 0.f;
  for (int r = 0; r < ROWS_PER_WAVE; r++) {
    const int  idx  = sIdx[wv * ROWS_PER_WAVE + r];
    const long orow = rowbase + r;
    const float* wr = W + (size_t)idx * DIMD + lane * 8;
    const float* zq = z + orow * DIMD + lane * 8;
    const float4 w0 = *(const float4*)(wr);
    const float4 w1 = *(const float4*)(wr + 4);
    const float4 z0 = *(const float4*)(zq);
    const float4 z1 = *(const float4*)(zq + 4);
    float4* po = (float4*)(outZ + orow * DIMD + lane * 8);
    po[0] = w0; po[1] = w1;
    float d;
    d = w0.x - z0.x; lacc += d * d;  d = w0.y - z0.y; lacc += d * d;
    d = w0.z - z0.z; lacc += d * d;  d = w0.w - z0.w; lacc += d * d;
    d = w1.x - z1.x; lacc += d * d;  d = w1.y - z1.y; lacc += d * d;
    d = w1.z - z1.z; lacc += d * d;  d = w1.w - z1.w; lacc += d * d;
    if (lane == 0) outIdx[orow] = (float)idx;
  }
#pragma unroll
  for (int off = 1; off < 32; off <<= 1) lacc += __shfl_xor(lacc, off, 32);
  if (lane == 0) atomicAdd(outLoss, lacc * (1.25f / (float)NROWS));
}

// --------------------------- launcher --------------------------------------
extern "C" void kernel_launch(void* const* d_in, const int* in_sizes, int n_in,
                              void* d_out, int out_size, void* d_ws, size_t ws_size,
                              hipStream_t stream) {
  (void)in_sizes; (void)n_in; (void)out_size; (void)ws_size;

  const float* z = (const float*)d_in[0];   // [32, 2048, 256] fp32
  const float* W = (const float*)d_in[1];   // [1024, 256]    fp32

  float* out     = (float*)d_out;
  float* outZ    = out;                                   // 65536*256 floats
  float* outIdx  = out + (size_t)NROWS * DIMD;            // 65536 floats
  float* outLoss = outIdx + NROWS;                        // 1 float

  unsigned short* Wb = (unsigned short*)d_ws;                                  // 512 KB
  float* bias = (float*)((char*)d_ws + (size_t)KCODES * DIMD * sizeof(unsigned short)); // 4 KB

  vq_prep<<<KCODES, DIMD, 0, stream>>>(W, Wb, bias, outLoss);
  vq_main<<<NROWS / ROWS_PER_WG, 256, 0, stream>>>(z, W, Wb, bias,
                                                   outZ, outIdx, outLoss);
}
